// NodeAttention_13932873908960
// MI455X (gfx1250) — matrix operand order repaired and checked by reading, not compiled, not run on hardware
//
#include <hip/hip_runtime.h>
#include <hip/hip_bf16.h>

typedef __attribute__((ext_vector_type(16))) _Float16 v16h;
typedef __attribute__((ext_vector_type(8)))  _Float16 v8h;
typedef __attribute__((ext_vector_type(8)))  float    v8f;

#define N_TOK 1024
#define D_NODE 256
#define D_EDGE 128
#define D_BIAS 64
#define D_HEAD 48
#define N_HEAD 8
#define D_QKV 384   // N_HEAD * D_HEAD
#define D_PAD 64    // padded head dim for WMMA
#define INFV 1000000.0f
#define EPSV 1e-5f

// ---------------------------------------------------------------------------
// Tiny setup: Wc = W_bias @ Wb  (64x128 @ 128x8 -> 64x8)
// ---------------------------------------------------------------------------
__global__ void wc_kernel(const float* __restrict__ Wbias,
                          const float* __restrict__ Wb,
                          float* __restrict__ Wc) {
  int t = threadIdx.x;
  if (t < D_BIAS * N_HEAD) {
    int c = t >> 3, h = t & 7;
    float s = 0.f;
    for (int d = 0; d < D_EDGE; ++d) s += Wbias[c * D_EDGE + d] * Wb[d * N_HEAD + h];
    Wc[t] = s;
  }
}

// ---------------------------------------------------------------------------
// Transpose + f32 -> f16 weight conversion: dst[n*K+k] = src[k*N+n]
// ---------------------------------------------------------------------------
__global__ void tconv_kernel(const float* __restrict__ src, _Float16* __restrict__ dst,
                             int K, int N) {
  int idx = blockIdx.x * 256 + threadIdx.x;
  if (idx < K * N) {
    int k = idx / N, n = idx % N;
    dst[(size_t)n * K + k] = (_Float16)src[idx];
  }
}

// ---------------------------------------------------------------------------
// b_term streaming kernel: one pass over edge (537MB) + bias (268MB).
// b_term[i,j,h] = sum_c LN(edge[i,j,:])[c]*Wb[c,h] + sum_c bias[i,j,c]*Wc[c,h]
// One wave per (i,j) pair; lane l owns 4 edge channels.
// ---------------------------------------------------------------------------
__global__ __launch_bounds__(256) void bterm_kernel(
    const float* __restrict__ edge, const float* __restrict__ bias,
    const float* __restrict__ g_edge, const float* __restrict__ b_edge,
    const float* __restrict__ Wb, const float* __restrict__ Wc,
    float* __restrict__ bt) {
  __shared__ float sWb[D_EDGE * N_HEAD];
  __shared__ float sWc[D_BIAS * N_HEAD];
  __shared__ float sG[D_EDGE], sB[D_EDGE];
  int tid = threadIdx.x;
  for (int e = tid; e < D_EDGE * N_HEAD; e += 256) sWb[e] = Wb[e];
  for (int e = tid; e < D_BIAS * N_HEAD; e += 256) sWc[e] = Wc[e];
  if (tid < D_EDGE) { sG[tid] = g_edge[tid]; sB[tid] = b_edge[tid]; }
  __syncthreads();

  int i = blockIdx.x;
  int w = tid >> 5, lane = tid & 31;
  for (int j = w; j < N_TOK; j += 8) {
    size_t pbase = (size_t)i * N_TOK + j;
    const float4 e4 = *(const float4*)(edge + pbase * D_EDGE + lane * 4);
    float ev[4] = {e4.x, e4.y, e4.z, e4.w};
    float s = ev[0] + ev[1] + ev[2] + ev[3];
    float sq = ev[0]*ev[0] + ev[1]*ev[1] + ev[2]*ev[2] + ev[3]*ev[3];
    #pragma unroll
    for (int o = 16; o >= 1; o >>= 1) {
      s  += __shfl_xor(s,  o, 32);
      sq += __shfl_xor(sq, o, 32);
    }
    float mean = s * (1.f / 128.f);
    float rstd = rsqrtf(sq * (1.f / 128.f) - mean * mean + EPSV);

    float acc[8] = {0.f,0.f,0.f,0.f,0.f,0.f,0.f,0.f};
    #pragma unroll
    for (int c4 = 0; c4 < 4; ++c4) {
      int c = lane * 4 + c4;
      float xn = (ev[c4] - mean) * rstd * sG[c] + sB[c];
      #pragma unroll
      for (int h = 0; h < 8; ++h) acc[h] += xn * sWb[c * 8 + h];
    }
    if (lane < 16) {
      const float4 b4 = *(const float4*)(bias + pbase * D_BIAS + lane * 4);
      float bv[4] = {b4.x, b4.y, b4.z, b4.w};
      #pragma unroll
      for (int c4 = 0; c4 < 4; ++c4) {
        int c = lane * 4 + c4;
        #pragma unroll
        for (int h = 0; h < 8; ++h) acc[h] += bv[c4] * sWc[c * 8 + h];
      }
    }
    #pragma unroll
    for (int h = 0; h < 8; ++h)
      #pragma unroll
      for (int o = 16; o >= 1; o >>= 1) acc[h] += __shfl_xor(acc[h], o, 32);
    if (lane == 0) {
      float4* op = (float4*)(bt + pbase * N_HEAD);
      op[0] = make_float4(acc[0], acc[1], acc[2], acc[3]);
      op[1] = make_float4(acc[4], acc[5], acc[6], acc[7]);
    }
  }
}

// ---------------------------------------------------------------------------
// Row LayerNorm (D=256) -> f16
// ---------------------------------------------------------------------------
__global__ __launch_bounds__(256) void ln_kernel(const float* __restrict__ x,
                                                 const float* __restrict__ g,
                                                 const float* __restrict__ b,
                                                 _Float16* __restrict__ out) {
  __shared__ float ps[8], pq[8];
  int row = blockIdx.x, tid = threadIdx.x;
  int w = tid >> 5, lane = tid & 31;
  float v = x[(size_t)row * D_NODE + tid];
  float s = v, sq = v * v;
  #pragma unroll
  for (int o = 16; o >= 1; o >>= 1) { s += __shfl_xor(s, o, 32); sq += __shfl_xor(sq, o, 32); }
  if (lane == 0) { ps[w] = s; pq[w] = sq; }
  __syncthreads();
  if (tid == 0) {
    float S = 0.f, Q = 0.f;
    for (int k = 0; k < 8; ++k) { S += ps[k]; Q += pq[k]; }
    float m = S * (1.f / D_NODE);
    ps[0] = m;
    pq[0] = rsqrtf(Q * (1.f / D_NODE) - m * m + EPSV);
  }
  __syncthreads();
  float mean = ps[0], rstd = pq[0];
  out[(size_t)row * D_NODE + tid] = (_Float16)((v - mean) * rstd * g[tid] + b[tid]);
}

// ---------------------------------------------------------------------------
// Generic WMMA GEMM: C[MxN] = A[MxK](f16,row) @ Bt[NxK](f16) with fusions.
// Block = 8 waves, each wave one 16x16 tile; block tile 64x32.
// Fragment layouts: A and B both k = 16*(lane/16) + s (contiguous), B col-per-lane.
// ---------------------------------------------------------------------------
template <int BIAS, int RELU, int RES, int O16, int O32>
__global__ __launch_bounds__(256) void gemm16_kernel(
    const _Float16* __restrict__ A, const _Float16* __restrict__ Bt,
    const float* __restrict__ bias, const float* __restrict__ res,
    float* __restrict__ of, _Float16* __restrict__ oh,
    int M, int N, int K) {
  int w = threadIdx.x >> 5, lane = threadIdx.x & 31;
  int m0 = blockIdx.x * 64 + (w & 3) * 16;
  int n0 = blockIdx.y * 32 + (w >> 2) * 16;
  int l16 = lane & 15;
  int aoff = (lane < 16) ? 0 : 16;
  int boff = (lane < 16) ? 0 : 16;
  v8f c = {};
  for (int kb = 0; kb < K; kb += 32) {
    v16h a, b;
    const _Float16* ap = A + (size_t)(m0 + l16) * K + kb + aoff;
    #pragma unroll
    for (int p = 0; p < 8; ++p) { a[p] = ap[p]; a[8 + p] = ap[8 + p]; }
    const _Float16* bp = Bt + (size_t)(n0 + l16) * K + kb + boff;
    #pragma unroll
    for (int p = 0; p < 16; ++p) b[p] = bp[p];
    c = __builtin_amdgcn_wmma_f32_16x16x32_f16(false, a, false, b, (short)0, c, false, false);
  }
  int nc = n0 + l16;
  int mb = m0 + ((lane < 16) ? 0 : 8);
  #pragma unroll
  for (int r = 0; r < 8; ++r) {
    int row = mb + r;
    float v = c[r];
    if (BIAS) v += bias[nc];
    if (RELU) v = v > 0.f ? v : 0.f;
    if (RES)  v += res[(size_t)row * N + nc];
    if (O32)  of[(size_t)row * N + nc] = v;
    if (O16)  oh[(size_t)row * N + nc] = (_Float16)v;
  }
}

// ---------------------------------------------------------------------------
// RoPE + scale + sigmoid-gate; emits padded head-major f16 q/k/v.
// q16/k16/v16 layout: [h][i][64] (d 48..63 zero-padded)
// ---------------------------------------------------------------------------
__global__ __launch_bounds__(384) void rope_gate_kernel(
    const float* __restrict__ qkvg, const float* __restrict__ node_pos,
    const float* __restrict__ bg, _Float16* __restrict__ q16,
    _Float16* __restrict__ k16, _Float16* __restrict__ v16,
    float* __restrict__ gate) {
  int i = blockIdx.x, t = threadIdx.x;     // t in [0,384)
  int h = t / D_HEAD, d = t % D_HEAD;
  float pos = node_pos[i];
  const float* qr = qkvg + (size_t)i * (4 * D_QKV);
  int tb = (d < 24) ? d : d - 24;
  float freq = __powf(10000.0f, -(float)tb / 24.0f);
  float th = pos * freq;
  float cs = __cosf(th), sn = __sinf(th);

  float x1 = qr[h * D_HEAD + tb],            x2 = qr[h * D_HEAD + tb + 24];
  float y1 = qr[D_QKV + h * D_HEAD + tb],    y2 = qr[D_QKV + h * D_HEAD + tb + 24];
  float qv = (d < 24) ? (x1 * cs - x2 * sn) : (x1 * sn + x2 * cs);
  float kv = (d < 24) ? (y1 * cs - y2 * sn) : (y1 * sn + y2 * cs);

  size_t hd = ((size_t)(h * N_TOK) + i) * D_PAD + d;
  q16[hd] = (_Float16)(qv * 0.0625f);        // scale = 1/sqrt(D_NODE) = 1/16
  k16[hd] = (_Float16)kv;
  v16[hd] = (_Float16)qr[2 * D_QKV + t];
  float g = qr[3 * D_QKV + t] + bg[t];
  gate[(size_t)i * D_QKV + t] = 1.0f / (1.0f + __expf(-g));
  if (t < N_HEAD * 16) {                     // zero pad d = 48..63
    int h2 = t >> 4, dp = D_HEAD + (t & 15);
    size_t hp = ((size_t)(h2 * N_TOK) + i) * D_PAD + dp;
    q16[hp] = (_Float16)0.f; k16[hp] = (_Float16)0.f; v16[hp] = (_Float16)0.f;
  }
}

// ---------------------------------------------------------------------------
// Attention: one block per (16-row i-tile, head). WMMA QK^T -> softmax(j)
// in LDS -> WMMA P@V, fused gate, f16 output [i][h*48+d].
// ---------------------------------------------------------------------------
#define SSTR 1032
__global__ __launch_bounds__(256) void attn_kernel(
    const _Float16* __restrict__ q16, const _Float16* __restrict__ k16,
    const _Float16* __restrict__ v16, const float* __restrict__ bt,
    const float* __restrict__ mask, const float* __restrict__ gate,
    _Float16* __restrict__ out16) {
  __shared__ _Float16 sQ[16 * D_PAD];
  __shared__ _Float16 sS[16 * SSTR];
  __shared__ float sMi[16];
  int h = blockIdx.y, i0 = blockIdx.x * 16;
  int tid = threadIdx.x, w = tid >> 5, lane = tid & 31;
  int l16 = lane & 15;
  int aoff = (lane < 16) ? 0 : 16;
  int boff = (lane < 16) ? 0 : 16;

  for (int e = tid; e < 16 * D_PAD; e += 256) {
    int r = e >> 6, d = e & 63;
    sQ[e] = q16[((size_t)(h * N_TOK) + i0 + r) * D_PAD + d];
  }
  if (tid < 16) sMi[tid] = mask[i0 + tid];
  __syncthreads();

  // S = q @ k^T + b_term + mask  (each wave covers 128 j-columns)
  for (int jt = 0; jt < 8; ++jt) {
    int j0 = w * 128 + jt * 16;
    v8f c = {};
    #pragma unroll
    for (int ks = 0; ks < 2; ++ks) {
      v16h a, b;
      int kb = ks * 32;
      const _Float16* ap = sQ + l16 * D_PAD + kb + aoff;
      #pragma unroll
      for (int p = 0; p < 8; ++p) { a[p] = ap[p]; a[8 + p] = ap[8 + p]; }
      const _Float16* bp = k16 + ((size_t)(h * N_TOK) + j0 + l16) * D_PAD + kb + boff;
      #pragma unroll
      for (int p = 0; p < 16; ++p) b[p] = bp[p];
      c = __builtin_amdgcn_wmma_f32_16x16x32_f16(false, a, false, b, (short)0, c, false, false);
    }
    int Mb = (lane < 16) ? 0 : 8;
    #pragma unroll
    for (int r = 0; r < 8; ++r) {
      int M = Mb + r, j = j0 + l16, i = i0 + M;
      float mv = INFV * (sMi[M] * mask[j] - 1.0f);
      float val = c[r] + bt[((size_t)i * N_TOK + j) * N_HEAD + h] + mv;
      sS[M * SSTR + j] = (_Float16)val;
    }
  }
  __syncthreads();

  // softmax over j: each wave owns 2 rows (16 lanes per row)
  {
    int row = w * 2 + (lane >> 4);
    float mx = -1e30f;
    for (int j = l16; j < N_TOK; j += 16) mx = fmaxf(mx, (float)sS[row * SSTR + j]);
    #pragma unroll
    for (int o = 8; o >= 1; o >>= 1) mx = fmaxf(mx, __shfl_xor(mx, o, 32));
    float sum = 0.f;
    for (int j = l16; j < N_TOK; j += 16) {
      float e = __expf((float)sS[row * SSTR + j] - mx);
      sum += e;
      sS[row * SSTR + j] = (_Float16)e;
    }
    #pragma unroll
    for (int o = 8; o >= 1; o >>= 1) sum += __shfl_xor(sum, o, 32);
    float inv = 1.0f / sum;
    for (int j = l16; j < N_TOK; j += 16)
      sS[row * SSTR + j] = (_Float16)((float)sS[row * SSTR + j] * inv);
  }
  __syncthreads();

  // O = P @ V (3 n-tiles of 16 cover d=0..47), fused gate
  if (w < 3) {
    v8f c = {};
    int d = w * 16 + l16;
    for (int kb = 0; kb < N_TOK; kb += 32) {
      v16h a, b;
      const _Float16* ap = sS + l16 * SSTR + kb + aoff;
      #pragma unroll
      for (int p = 0; p < 8; ++p) { a[p] = ap[p]; a[8 + p] = ap[8 + p]; }
      const _Float16* vp = v16 + ((size_t)(h * N_TOK) + kb + boff) * D_PAD + d;
      #pragma unroll
      for (int p = 0; p < 16; ++p) b[p] = vp[(size_t)p * D_PAD];
      c = __builtin_amdgcn_wmma_f32_16x16x32_f16(false, a, false, b, (short)0, c, false, false);
    }
    int Mb = (lane < 16) ? 0 : 8;
    int col = h * D_HEAD + d;
    #pragma unroll
    for (int r = 0; r < 8; ++r) {
      int i = i0 + Mb + r;
      float g = gate[(size_t)i * D_QKV + col];
      out16[(size_t)i * D_QKV + col] = (_Float16)(c[r] * g);
    }
  }
}

// ---------------------------------------------------------------------------
// Host launch
// ---------------------------------------------------------------------------
extern "C" void kernel_launch(void* const* d_in, const int* in_sizes, int n_in,
                              void* d_out, int out_size, void* d_ws, size_t ws_size,
                              hipStream_t stream) {
  const float* node     = (const float*)d_in[0];
  const float* edge     = (const float*)d_in[1];
  const float* biasT    = (const float*)d_in[2];
  const float* node_pos = (const float*)d_in[3];
  const float* mask     = (const float*)d_in[4];
  const float* g_node   = (const float*)d_in[5];
  const float* b_node   = (const float*)d_in[6];
  const float* g_edge   = (const float*)d_in[7];
  const float* b_edge   = (const float*)d_in[8];
  const float* W_bias   = (const float*)d_in[9];
  const float* Wq       = (const float*)d_in[10];
  const float* Wk       = (const float*)d_in[11];
  const float* Wv       = (const float*)d_in[12];
  const float* Wb       = (const float*)d_in[13];
  const float* Wg       = (const float*)d_in[14];
  const float* bg       = (const float*)d_in[15];
  const float* Wo       = (const float*)d_in[16];
  const float* g_ff     = (const float*)d_in[17];
  const float* b_ff     = (const float*)d_in[18];
  const float* W1       = (const float*)d_in[19];
  const float* b1       = (const float*)d_in[20];
  const float* W2       = (const float*)d_in[21];
  const float* b2       = (const float*)d_in[22];
  (void)in_sizes; (void)n_in; (void)out_size; (void)ws_size;

  char* ws = (char*)d_ws;
  size_t off = 0;
  auto alloc = [&](size_t bytes) -> void* {
    off = (off + 255) & ~(size_t)255;
    void* p = ws + off;
    off += bytes;
    return p;
  };

  float*     bt        = (float*)alloc((size_t)N_TOK * N_TOK * N_HEAD * 4);
  float*     Wc        = (float*)alloc(D_BIAS * N_HEAD * 4);
  _Float16*  wqkvg_t   = (_Float16*)alloc((size_t)4 * D_QKV * D_NODE * 2);
  _Float16*  wo_t      = (_Float16*)alloc((size_t)D_NODE * D_QKV * 2);
  _Float16*  w1_t      = (_Float16*)alloc((size_t)(2 * D_NODE) * D_NODE * 2);
  _Float16*  w2_t      = (_Float16*)alloc((size_t)D_NODE * (2 * D_NODE) * 2);
  _Float16*  node_n16  = (_Float16*)alloc((size_t)N_TOK * D_NODE * 2);
  float*     qkvg      = (float*)alloc((size_t)N_TOK * 4 * D_QKV * 4);
  _Float16*  q16       = (_Float16*)alloc((size_t)N_HEAD * N_TOK * D_PAD * 2);
  _Float16*  k16       = (_Float16*)alloc((size_t)N_HEAD * N_TOK * D_PAD * 2);
  _Float16*  v16       = (_Float16*)alloc((size_t)N_HEAD * N_TOK * D_PAD * 2);
  float*     gateb     = (float*)alloc((size_t)N_TOK * D_QKV * 4);
  _Float16*  attn_o16  = (_Float16*)alloc((size_t)N_TOK * D_QKV * 2);
  float*     attn_lin  = (float*)alloc((size_t)N_TOK * D_NODE * 4);
  _Float16*  ln_ff16   = (_Float16*)alloc((size_t)N_TOK * D_NODE * 2);
  _Float16*  h1_16     = (_Float16*)alloc((size_t)N_TOK * 2 * D_NODE * 2);

  // setup: combined bias weight + f16 transposed weights
  wc_kernel<<<1, 512, 0, stream>>>(W_bias, Wb, Wc);
  auto tc = [&](const float* src, _Float16* dst, int K, int N) {
    tconv_kernel<<<(K * N + 255) / 256, 256, 0, stream>>>(src, dst, K, N);
  };
  tc(Wq, wqkvg_t + (size_t)0 * D_QKV * D_NODE, D_NODE, D_QKV);
  tc(Wk, wqkvg_t + (size_t)1 * D_QKV * D_NODE, D_NODE, D_QKV);
  tc(Wv, wqkvg_t + (size_t)2 * D_QKV * D_NODE, D_NODE, D_QKV);
  tc(Wg, wqkvg_t + (size_t)3 * D_QKV * D_NODE, D_NODE, D_QKV);
  tc(Wo, wo_t, D_QKV, D_NODE);
  tc(W1, w1_t, D_NODE, 2 * D_NODE);
  tc(W2, w2_t, 2 * D_NODE, D_NODE);

  // fused LN(edge)@Wb + bias@Wc  (the bandwidth-dominant pass)
  bterm_kernel<<<N_TOK, 256, 0, stream>>>(edge, biasT, g_edge, b_edge, Wb, Wc, bt);

  // node LN -> f16, then fused QKVG projection GEMM (1024 x 256 x 1536)
  ln_kernel<<<N_TOK, 256, 0, stream>>>(node, g_node, b_node, node_n16);
  gemm16_kernel<0, 0, 0, 0, 1><<<dim3(N_TOK / 64, (4 * D_QKV) / 32), 256, 0, stream>>>(
      node_n16, wqkvg_t, nullptr, nullptr, qkvg, nullptr, N_TOK, 4 * D_QKV, D_NODE);

  // rope + scale + gate, padded head-major f16 q/k/v
  rope_gate_kernel<<<N_TOK, 384, 0, stream>>>(qkvg, node_pos, bg, q16, k16, v16, gateb);

  // attention (64 i-tiles x 8 heads)
  attn_kernel<<<dim3(N_TOK / 16, N_HEAD), 256, 0, stream>>>(
      q16, k16, v16, bt, mask, gateb, attn_o16);

  // Wo projection (1024 x 384 x 256)
  gemm16_kernel<0, 0, 0, 0, 1><<<dim3(N_TOK / 64, D_NODE / 32), 256, 0, stream>>>(
      attn_o16, wo_t, nullptr, nullptr, attn_lin, nullptr, N_TOK, D_NODE, D_QKV);

  // FFN: LN -> W1+b1+relu (f16) -> W2+b2 + residual(node) -> d_out
  ln_kernel<<<N_TOK, 256, 0, stream>>>(attn_lin, g_ff, b_ff, ln_ff16);
  gemm16_kernel<1, 1, 0, 1, 0><<<dim3(N_TOK / 64, (2 * D_NODE) / 32), 256, 0, stream>>>(
      ln_ff16, w1_t, b1, nullptr, nullptr, h1_16, N_TOK, 2 * D_NODE, D_NODE);
  gemm16_kernel<1, 0, 1, 0, 1><<<dim3(N_TOK / 64, D_NODE / 32), 256, 0, stream>>>(
      h1_16, w2_t, b2, node, (float*)d_out, nullptr, N_TOK, D_NODE, 2 * D_NODE);
}
